// SlotAttention_18674517803644
// MI455X (gfx1250) — matrix operand order repaired and checked by reading, not compiled, not run on hardware
//
#include <hip/hip_runtime.h>
#include <hip/hip_bf16.h>

// ---------------------------------------------------------------------------
// Slot Attention for MI455X (gfx1250, wave32, WMMA + TDM/async-LDS).
// B=64 L=4096 DIN=D=256 N=8 ITERS=5.
// Big GEMMs in bf16 WMMA (v_wmma_f32_16x16x32_bf16); small per-slot math f32.
// q-tile staged to LDS with tensor_load_to_lds (TDM); row staging in the
// finalize kernel uses global_load_async_to_lds_b32 (+ s_wait_asynccnt).
// ---------------------------------------------------------------------------

typedef __bf16 bf16;
typedef __attribute__((ext_vector_type(16))) __bf16 bf16x16;
typedef __attribute__((ext_vector_type(8)))  __bf16 bf16x8;
typedef __attribute__((ext_vector_type(8)))  float  f32x8;
typedef unsigned int u32x4 __attribute__((ext_vector_type(4)));
typedef int          i32x4 __attribute__((ext_vector_type(4)));
typedef int          i32x8 __attribute__((ext_vector_type(8)));

#define DMODEL   256
#define NSLOTS   8
#define BBATCH   64
#define LSEQ     4096
#define NITERS   5
#define MLPH     128
#define SCALE_QK 0.0625f   /* 256^-0.5 */
#define ATTN_EPS 1e-8f
#define LN_EPSV  1e-5f
#define LCHUNK   512       /* l-rows per attention block */
#define NCHUNK   (LSEQ / LCHUNK)

__device__ __forceinline__ f32x8 zero8() {
  f32x8 z;
#pragma unroll
  for (int i = 0; i < 8; ++i) z[i] = 0.0f;
  return z;
}

__device__ __forceinline__ f32x8 wmma_bf16(bf16x16 a, bf16x16 b, f32x8 c) {
  return __builtin_amdgcn_wmma_f32_16x16x32_bf16(false, a, false, b,
                                                 (short)0, c, false, false);
}

// A fragment: 16(M) x 32(K) bf16 from a row-major [M x ld] tile.
// ISA layout: lane<16 -> row=lane, K={kb..kb+7, kb+16..kb+23};
//             lane>=16 -> row=lane-16, K={kb+8..kb+15, kb+24..kb+31}.
__device__ __forceinline__ bf16x16 load_a_frag(const bf16* base, int ld,
                                               int kBase, int lane) {
  const int r  = lane & 15;
  const int k0 = kBase + ((lane & 16) ? 8 : 0);
  const bf16* p = base + r * ld + k0;
  bf16x8 lo = *(const bf16x8*)(p);
  bf16x8 hi = *(const bf16x8*)(p + 16);
  bf16x16 a;
#pragma unroll
  for (int i = 0; i < 8; ++i) { a[i] = lo[i]; a[i + 8] = hi[i]; }
  return a;
}

// B fragment: 32(K) x 16(N), element (k,n) at base[n*ldn + k] (K contiguous).
// ISA layout: lane<16 -> col=lane, K={kb..kb+7, kb+16..kb+23};
//             lane>=16 -> col=lane-16, K={kb+8..kb+15, kb+24..kb+31}.
__device__ __forceinline__ bf16x16 load_b_frag(const bf16* base, int ldn,
                                               int kBase, int lane) {
  const int n  = lane & 15;
  const int k0 = kBase + ((lane & 16) ? 8 : 0);
  const bf16* p = base + n * ldn + k0;
  bf16x8 lo = *(const bf16x8*)(p);
  bf16x8 hi = *(const bf16x8*)(p + 16);
  bf16x16 b;
#pragma unroll
  for (int i = 0; i < 8; ++i) { b[i] = lo[i]; b[i + 8] = hi[i]; }
  return b;
}

// Async per-lane 4B copy global -> LDS (GLOBAL_LOAD_ASYNC_TO_LDS_B32, GV mode).
__device__ __forceinline__ void async_g2l_b32(unsigned lds_off,
                                              unsigned long long gaddr) {
  asm volatile("global_load_async_to_lds_b32 %0, %1, off" ::"v"(lds_off),
               "v"(gaddr)
               : "memory");
}
__device__ __forceinline__ void wait_asynccnt0() {
  asm volatile("s_wait_asynccnt 0" ::: "memory");
}

// --------------------------- utility kernels -------------------------------

__global__ void k_f32_to_bf16(const float* __restrict__ src,
                              bf16* __restrict__ dst, int n) {
  int i = blockIdx.x * blockDim.x + threadIdx.x;
  if (i < n) dst[i] = (bf16)src[i];
}

__global__ void k_zero(float* __restrict__ p, int n) {
  int i = blockIdx.x * blockDim.x + threadIdx.x;
  if (i < n) p[i] = 0.0f;
}

__global__ void k_init_slots(const float* __restrict__ noise,
                             const float* __restrict__ mu,
                             const float* __restrict__ logsig,
                             float* __restrict__ slots, int n) {
  int i = blockIdx.x * blockDim.x + threadIdx.x;
  if (i < n) {
    int d = i & (DMODEL - 1);
    slots[i] = mu[d] + __expf(logsig[d]) * noise[i];
  }
}

// ---------------- fused LayerNorm + K/V projection (bf16 WMMA) -------------
// Grid: (B*L)/16 blocks, 256 threads (8 waves). Each block: 16 rows.
// 8 waves x 4 tiles (2 K-cols + 2 V-cols of 16) cover the 2x256 outputs.
__global__ __launch_bounds__(256) void k_ln_proj_kv(
    const float* __restrict__ in, const float* __restrict__ g,
    const float* __restrict__ bta, const bf16* __restrict__ WkB,
    const bf16* __restrict__ WvB, bf16* __restrict__ K, bf16* __restrict__ V) {
  __shared__ __align__(16) bf16 xs[16 * DMODEL];  // 8 KB LN'd inputs
  const int m0   = blockIdx.x * 16;
  const int wave = threadIdx.x >> 5;
  const int lane = threadIdx.x & 31;

#pragma unroll
  for (int rr = 0; rr < 2; ++rr) {
    const int r = wave * 2 + rr;
    const float* row = in + (size_t)(m0 + r) * DMODEL;
    float vals[8], s = 0.f, s2 = 0.f;
#pragma unroll
    for (int i = 0; i < 8; ++i) {
      float x = row[lane + 32 * i];
      vals[i] = x; s += x; s2 += x * x;
    }
#pragma unroll
    for (int m = 16; m >= 1; m >>= 1) {
      s  += __shfl_xor(s, m, 32);
      s2 += __shfl_xor(s2, m, 32);
    }
    const float mean = s * (1.0f / DMODEL);
    const float var  = s2 * (1.0f / DMODEL) - mean * mean;
    const float rstd = rsqrtf(var + LN_EPSV);
#pragma unroll
    for (int i = 0; i < 8; ++i) {
      const int d = lane + 32 * i;
      xs[r * DMODEL + d] = (bf16)((vals[i] - mean) * rstd * g[d] + bta[d]);
    }
  }
  __syncthreads();

  f32x8 acc[4];
#pragma unroll
  for (int t = 0; t < 4; ++t) acc[t] = zero8();

  for (int kb = 0; kb < DMODEL; kb += 32) {
    bf16x16 a = load_a_frag(xs, DMODEL, kb, lane);
#pragma unroll
    for (int t = 0; t < 4; ++t) {
      const bf16* W = (t < 2) ? WkB : WvB;
      const int n0 = (wave * 2 + (t & 1)) * 16;
      bf16x16 bf = load_b_frag(W + n0 * DMODEL, DMODEL, kb, lane);
      acc[t] = wmma_bf16(a, bf, acc[t]);
    }
  }
#pragma unroll
  for (int t = 0; t < 4; ++t) {
    bf16* O = (t < 2) ? K : V;
    const int n0    = (wave * 2 + (t & 1)) * 16;
    const int col   = n0 + (lane & 15);
    const int rbase = (lane & 16) ? 8 : 0;
#pragma unroll
    for (int r = 0; r < 8; ++r)
      O[(size_t)(m0 + rbase + r) * DMODEL + col] = (bf16)acc[t][r];
  }
}

// ------------- slot LayerNorm + Q projection (bf16 WMMA) -------------------
__global__ __launch_bounds__(256) void k_ln_proj_q(
    const float* __restrict__ slots, const float* __restrict__ g,
    const float* __restrict__ bta, const bf16* __restrict__ WqB,
    bf16* __restrict__ Q) {
  __shared__ __align__(16) bf16 xs[16 * DMODEL];
  const int m0   = blockIdx.x * 16;
  const int wave = threadIdx.x >> 5;
  const int lane = threadIdx.x & 31;
#pragma unroll
  for (int rr = 0; rr < 2; ++rr) {
    const int r = wave * 2 + rr;
    const float* row = slots + (size_t)(m0 + r) * DMODEL;
    float vals[8], s = 0.f, s2 = 0.f;
#pragma unroll
    for (int i = 0; i < 8; ++i) {
      float x = row[lane + 32 * i];
      vals[i] = x; s += x; s2 += x * x;
    }
#pragma unroll
    for (int m = 16; m >= 1; m >>= 1) {
      s  += __shfl_xor(s, m, 32);
      s2 += __shfl_xor(s2, m, 32);
    }
    const float mean = s * (1.0f / DMODEL);
    const float var  = s2 * (1.0f / DMODEL) - mean * mean;
    const float rstd = rsqrtf(var + LN_EPSV);
#pragma unroll
    for (int i = 0; i < 8; ++i) {
      const int d = lane + 32 * i;
      xs[r * DMODEL + d] = (bf16)((vals[i] - mean) * rstd * g[d] + bta[d]);
    }
  }
  __syncthreads();

  f32x8 acc[2];
  acc[0] = zero8(); acc[1] = zero8();
  for (int kb = 0; kb < DMODEL; kb += 32) {
    bf16x16 a = load_a_frag(xs, DMODEL, kb, lane);
#pragma unroll
    for (int t = 0; t < 2; ++t) {
      const int n0 = (wave * 2 + t) * 16;
      bf16x16 bf = load_b_frag(WqB + n0 * DMODEL, DMODEL, kb, lane);
      acc[t] = wmma_bf16(a, bf, acc[t]);
    }
  }
#pragma unroll
  for (int t = 0; t < 2; ++t) {
    const int n0    = (wave * 2 + t) * 16;
    const int col   = n0 + (lane & 15);
    const int rbase = (lane & 16) ? 8 : 0;
#pragma unroll
    for (int r = 0; r < 8; ++r)
      Q[(size_t)(m0 + rbase + r) * DMODEL + col] = (bf16)acc[t][r];
  }
}

// ----------------------- attention core (bf16 WMMA) ------------------------
// Softmax over the slot axis lives in one lane's 8 accumulator VGPRs after
// the dots WMMA. L-renorm commutes with attn@v: accumulate U = sum_l a*v and
// S = sum_l a (a = softmax + EPS) with f32 atomics.
// Grid: (NCHUNK, B), 256 threads (8 waves), wave owns 64 l (2 super-tiles).
__global__ __launch_bounds__(256) void k_attn(
    const bf16* __restrict__ Q, const bf16* __restrict__ K,
    const bf16* __restrict__ V, float* __restrict__ U, float* __restrict__ S) {
  const int b     = blockIdx.y;
  const int chunk = blockIdx.x;
  const int wave  = threadIdx.x >> 5;
  const int lane  = threadIdx.x & 31;

  __shared__ __align__(16) bf16 qs[16 * DMODEL];   // 8 KB (rows 8..15 zero)
  __shared__ __align__(16) bf16 alds[8][16 * 32];  // 8 KB attn A-tiles
  __shared__ __align__(16) bf16 vt[8][64 * 32];    // 32 KB v^T staging

  // --- stage q tile (8x256 bf16, contiguous 4KB) into LDS via TDM ---------
#if __has_builtin(__builtin_amdgcn_tensor_load_to_lds)
  if (wave == 0) {
    const unsigned qlds = (unsigned)(size_t)(void*)qs;
    const unsigned long long ga =
        (unsigned long long)(size_t)(Q + (size_t)b * NSLOTS * DMODEL);
    u32x4 g0;
    g0[0] = 1u;                                   // count=1 (user mode)
    g0[1] = qlds;                                 // lds_addr
    g0[2] = (unsigned)(ga & 0xffffffffu);         // global_addr[31:0]
    g0[3] = (unsigned)((ga >> 32) & 0x01ffffffu)  // global_addr[56:32]
            | (2u << 30);                         // type = 2 ("image")
    i32x8 g1;
    g1[0] = (1 << 16);                 // data_size = 1 -> 2 bytes
    g1[1] = (int)(256u << 16);         // tensor_dim0 = 256 (low 16)
    g1[2] = (int)(8u << 16);           // tensor_dim0 hi=0 | tensor_dim1 = 8
    g1[3] = (int)(256u << 16);         // tensor_dim1 hi=0 | tile_dim0 = 256
    g1[4] = 8;                         // tile_dim1 = 8, tile_dim2 = 0
    g1[5] = 256;                       // tensor_dim0_stride (low 32)
    g1[6] = 0;                         // stride hi | tensor_dim1_stride lo
    g1[7] = 0;
    i32x4 z4;
    z4[0] = 0; z4[1] = 0; z4[2] = 0; z4[3] = 0;
#if defined(__clang_major__) && (__clang_major__ >= 23)
    i32x8 z8;
#pragma unroll
    for (int i = 0; i < 8; ++i) z8[i] = 0;
    __builtin_amdgcn_tensor_load_to_lds(g0, g1, z4, z4, z8, 0);
#else
    __builtin_amdgcn_tensor_load_to_lds(g0, g1, z4, z4, 0);
#endif
  }
  // zero pad rows 8..15 while the TDM runs
  for (int i = threadIdx.x; i < NSLOTS * DMODEL; i += 256)
    qs[NSLOTS * DMODEL + i] = (bf16)0.f;
#if __has_builtin(__builtin_amdgcn_s_wait_tensorcnt)
  __builtin_amdgcn_s_wait_tensorcnt((short)0);
#else
  asm volatile("s_wait_tensorcnt 0" ::: "memory");
#endif
#else
  for (int i = threadIdx.x; i < 16 * DMODEL; i += 256) {
    int r = i >> 8, d = i & (DMODEL - 1);
    qs[i] = (r < NSLOTS) ? Q[((size_t)b * NSLOTS + r) * DMODEL + d] : (bf16)0.f;
  }
#endif
  __syncthreads();

  const size_t kvbase = (size_t)b * LSEQ * DMODEL;
  float s_acc[8];
#pragma unroll
  for (int r = 0; r < 8; ++r) s_acc[r] = 0.0f;
  f32x8 uacc[16];
#pragma unroll
  for (int t = 0; t < 16; ++t) uacc[t] = zero8();

#pragma unroll
  for (int st = 0; st < 2; ++st) {
    const int l0 = chunk * LCHUNK + wave * 64 + st * 32;
    __builtin_prefetch(K + kvbase + (size_t)(l0 + 32) * DMODEL, 0, 1);
    __builtin_prefetch(V + kvbase + (size_t)l0 * DMODEL, 0, 1);

    // dots[n, l] for 32 l-columns: two 16x16 WMMA chains over K=256.
    f32x8 dots[2];
    dots[0] = zero8(); dots[1] = zero8();
    for (int kb = 0; kb < DMODEL; kb += 32) {
      bf16x16 a = load_a_frag(qs, DMODEL, kb, lane);
#pragma unroll
      for (int h = 0; h < 2; ++h) {
        bf16x16 bf =
            load_b_frag(K + kvbase + (size_t)(l0 + h * 16) * DMODEL, DMODEL,
                        kb, lane);
        dots[h] = wmma_bf16(a, bf, dots[h]);
      }
    }

    // Per-column softmax over the 8 slots (lanes 0..15 hold valid rows).
    const bool act = (lane < 16);
#pragma unroll
    for (int h = 0; h < 2; ++h) {
      float v[8], m = -1e30f;
#pragma unroll
      for (int r = 0; r < 8; ++r) {
        v[r] = dots[h][r] * SCALE_QK;
        m = fmaxf(m, v[r]);
      }
      float e[8], sum = 0.0f;
#pragma unroll
      for (int r = 0; r < 8; ++r) { e[r] = __expf(v[r] - m); sum += e[r]; }
      const float inv = 1.0f / sum;
#pragma unroll
      for (int r = 0; r < 8; ++r) {
        const float aval = act ? (e[r] * inv + ATTN_EPS) : 0.0f;
        s_acc[r] += aval;  // lanes >=16 add 0
        const int row = r + ((lane & 16) ? 8 : 0);
        alds[wave][row * 32 + h * 16 + (lane & 15)] = (bf16)aval;
      }
    }

    // updates += attn(16x32) @ v(32x256), v^T staged in LDS in d-quarters.
    // lane == l-row: 8x global_load_b128 per quarter, b16 scatter stores.
    bf16x16 afrag = load_a_frag(alds[wave], 32, 0, lane);
    const bf16* vrow = V + kvbase + (size_t)(l0 + lane) * DMODEL;
#pragma unroll
    for (int qd = 0; qd < 4; ++qd) {
#pragma unroll
      for (int c = 0; c < 8; ++c) {
        bf16x8 vv = *(const bf16x8*)(vrow + qd * 64 + c * 8);
#pragma unroll
        for (int i = 0; i < 8; ++i)
          vt[wave][(c * 8 + i) * 32 + lane] = vv[i];
      }
#pragma unroll
      for (int dt = 0; dt < 4; ++dt) {
        bf16x16 bfr = load_b_frag(&vt[wave][dt * 16 * 32], 32, 0, lane);
        uacc[qd * 4 + dt] = wmma_bf16(afrag, bfr, uacc[qd * 4 + dt]);
      }
    }
  }

  // Flush: U rows n=0..7 live in lanes 0..15 (VGPR r == n).
  if (lane < 16) {
#pragma unroll
    for (int dt = 0; dt < 16; ++dt)
#pragma unroll
      for (int r = 0; r < 8; ++r)
        atomicAdd(&U[((size_t)b * NSLOTS + r) * DMODEL + dt * 16 + lane],
                  uacc[dt][r]);
  }
#pragma unroll
  for (int r = 0; r < 8; ++r) {
    float v = s_acc[r];
    v += __shfl_xor(v, 1, 32);
    v += __shfl_xor(v, 2, 32);
    v += __shfl_xor(v, 4, 32);
    v += __shfl_xor(v, 8, 32);
    if (lane == 0) atomicAdd(&S[b * NSLOTS + r], v);
  }
}

// ----------------- GRU + LayerNorm + MLP finalize (f32 VALU) ---------------
// ~0.3 GFLOP/iter. Grid: 512 blocks (one slot row), 256 threads. Row staging
// uses async global->LDS copies (ASYNCcnt path).
__global__ __launch_bounds__(256) void k_update(
    const float* __restrict__ U, const float* __restrict__ S,
    const float* __restrict__ slots_prev, const float* __restrict__ w_ih,
    const float* __restrict__ w_hh, const float* __restrict__ b_ih,
    const float* __restrict__ b_hh, const float* __restrict__ g_ff,
    const float* __restrict__ bt_ff, const float* __restrict__ mw1,
    const float* __restrict__ mb1, const float* __restrict__ mw2,
    const float* __restrict__ mb2, float* __restrict__ slots_out) {
  const int row = blockIdx.x;
  const int d   = threadIdx.x;
  __shared__ float upd[DMODEL], hprev[DMODEL], ffs[DMODEL], ff1[MLPH];
  __shared__ float ws1[8], ws2[8];

  const float sinv = 1.0f / S[row];
  // Async per-lane copies of this row's updates and previous slot state.
  async_g2l_b32((unsigned)(size_t)(void*)&upd[d],
                (unsigned long long)(size_t)(U + (size_t)row * DMODEL + d));
  async_g2l_b32(
      (unsigned)(size_t)(void*)&hprev[d],
      (unsigned long long)(size_t)(slots_prev + (size_t)row * DMODEL + d));
  wait_asynccnt0();
  __syncthreads();
  upd[d] *= sinv;  // renormalize in LDS
  __syncthreads();

  float gxr = b_ih[d], gxz = b_ih[DMODEL + d], gxn = b_ih[2 * DMODEL + d];
  float ghr = b_hh[d], ghz = b_hh[DMODEL + d], ghn = b_hh[2 * DMODEL + d];
  for (int k = 0; k < DMODEL; ++k) {
    const float u = upd[k], h = hprev[k];
    gxr += u * w_ih[(size_t)d * DMODEL + k];
    gxz += u * w_ih[(size_t)(DMODEL + d) * DMODEL + k];
    gxn += u * w_ih[(size_t)(2 * DMODEL + d) * DMODEL + k];
    ghr += h * w_hh[(size_t)d * DMODEL + k];
    ghz += h * w_hh[(size_t)(DMODEL + d) * DMODEL + k];
    ghn += h * w_hh[(size_t)(2 * DMODEL + d) * DMODEL + k];
  }
  const float rg = 1.0f / (1.0f + __expf(-(gxr + ghr)));
  const float zg = 1.0f / (1.0f + __expf(-(gxz + ghz)));
  const float ng = tanhf(gxn + rg * ghn);
  const float h  = (1.0f - zg) * ng + zg * hprev[d];

  float s1 = h, s2 = h * h;
#pragma unroll
  for (int m = 16; m >= 1; m >>= 1) {
    s1 += __shfl_xor(s1, m, 32);
    s2 += __shfl_xor(s2, m, 32);
  }
  if ((threadIdx.x & 31) == 0) {
    ws1[threadIdx.x >> 5] = s1;
    ws2[threadIdx.x >> 5] = s2;
  }
  __syncthreads();
  float t1 = 0.f, t2 = 0.f;
#pragma unroll
  for (int w = 0; w < 8; ++w) { t1 += ws1[w]; t2 += ws2[w]; }
  const float mean = t1 * (1.0f / DMODEL);
  const float var  = t2 * (1.0f / DMODEL) - mean * mean;
  const float rstd = rsqrtf(var + LN_EPSV);
  ffs[d] = (h - mean) * rstd * g_ff[d] + bt_ff[d];
  __syncthreads();

  if (d < MLPH) {
    float acc = mb1[d];
    for (int k = 0; k < DMODEL; ++k) acc += ffs[k] * mw1[(size_t)d * DMODEL + k];
    ff1[d] = fmaxf(acc, 0.0f);
  }
  __syncthreads();
  float acc2 = mb2[d];
  for (int k = 0; k < MLPH; ++k) acc2 += ff1[k] * mw2[(size_t)d * MLPH + k];
  slots_out[(size_t)row * DMODEL + d] = h + acc2;
}

// ------------------------------- launcher ----------------------------------

extern "C" void kernel_launch(void* const* d_in, const int* in_sizes, int n_in,
                              void* d_out, int out_size, void* d_ws,
                              size_t ws_size, hipStream_t stream) {
  const float* inputs  = (const float*)d_in[0];
  const float* noise   = (const float*)d_in[1];
  const float* mu      = (const float*)d_in[2];
  const float* logsig  = (const float*)d_in[3];
  const float* ln_in_g = (const float*)d_in[4];
  const float* ln_in_b = (const float*)d_in[5];
  const float* ln_s_g  = (const float*)d_in[6];
  const float* ln_s_b  = (const float*)d_in[7];
  const float* ln_ff_g = (const float*)d_in[8];
  const float* ln_ff_b = (const float*)d_in[9];
  const float* Wk      = (const float*)d_in[10];
  const float* Wv      = (const float*)d_in[11];
  const float* Wq      = (const float*)d_in[12];
  const float* w_ih    = (const float*)d_in[13];
  const float* w_hh    = (const float*)d_in[14];
  const float* b_ih    = (const float*)d_in[15];
  const float* b_hh    = (const float*)d_in[16];
  const float* mw1     = (const float*)d_in[17];
  const float* mb1     = (const float*)d_in[18];
  const float* mw2     = (const float*)d_in[19];
  const float* mb2     = (const float*)d_in[20];

  char* base = (char*)d_ws;
  size_t off = 0;
  auto take = [&](size_t bytes) -> char* {
    char* p = base + off;
    off = (off + bytes + 255) & ~(size_t)255;
    return p;
  };
  const size_t kvBytes = (size_t)BBATCH * LSEQ * DMODEL * sizeof(bf16);
  bf16*  Kbf    = (bf16*)take(kvBytes);
  bf16*  Vbf    = (bf16*)take(kvBytes);
  bf16*  WkB    = (bf16*)take(DMODEL * DMODEL * sizeof(bf16));
  bf16*  WvB    = (bf16*)take(DMODEL * DMODEL * sizeof(bf16));
  bf16*  WqB    = (bf16*)take(DMODEL * DMODEL * sizeof(bf16));
  bf16*  Qbf    = (bf16*)take((size_t)BBATCH * NSLOTS * DMODEL * sizeof(bf16));
  float* Uacc   = (float*)take((size_t)BBATCH * NSLOTS * DMODEL * sizeof(float));
  float* Sacc   = (float*)take((size_t)BBATCH * NSLOTS * sizeof(float));
  float* slotsA = (float*)take((size_t)BBATCH * NSLOTS * DMODEL * sizeof(float));
  float* slotsB = (float*)take((size_t)BBATCH * NSLOTS * DMODEL * sizeof(float));
  (void)ws_size; (void)in_sizes; (void)n_in; (void)out_size;

  const int wn = DMODEL * DMODEL;
  k_f32_to_bf16<<<(wn + 255) / 256, 256, 0, stream>>>(Wk, WkB, wn);
  k_f32_to_bf16<<<(wn + 255) / 256, 256, 0, stream>>>(Wv, WvB, wn);
  k_f32_to_bf16<<<(wn + 255) / 256, 256, 0, stream>>>(Wq, WqB, wn);

  const int sn = BBATCH * NSLOTS * DMODEL;
  k_init_slots<<<(sn + 255) / 256, 256, 0, stream>>>(noise, mu, logsig, slotsA,
                                                     sn);

  k_ln_proj_kv<<<(BBATCH * LSEQ) / 16, 256, 0, stream>>>(
      inputs, ln_in_g, ln_in_b, WkB, WvB, Kbf, Vbf);

  float* cur = slotsA;
  float* nxt = slotsB;
  for (int it = 0; it < NITERS; ++it) {
    k_ln_proj_q<<<(BBATCH * NSLOTS) / 16, 256, 0, stream>>>(cur, ln_s_g,
                                                            ln_s_b, WqB, Qbf);
    const int zn = BBATCH * NSLOTS * DMODEL + BBATCH * NSLOTS;
    k_zero<<<(zn + 255) / 256, 256, 0, stream>>>(Uacc, zn);
    k_attn<<<dim3(NCHUNK, BBATCH), 256, 0, stream>>>(Qbf, Kbf, Vbf, Uacc,
                                                     Sacc);
    float* outp = (it == NITERS - 1) ? (float*)d_out : nxt;
    k_update<<<BBATCH * NSLOTS, 256, 0, stream>>>(
        Uacc, Sacc, cur, w_ih, w_hh, b_ih, b_hh, ln_ff_g, ln_ff_b, mw1, mb1,
        mw2, mb2, outp);
    float* t = cur; cur = nxt; nxt = t;
  }
}
